// MEGNetRL_39883066311301
// MI455X (gfx1250) — matrix operand order repaired and checked by reading, not compiled, not run on hardware
//
#include <hip/hip_runtime.h>
#include <math.h>

typedef _Float16 f16;
typedef _Float16 v16h __attribute__((ext_vector_type(16)));
typedef _Float16 h8   __attribute__((ext_vector_type(8)));
typedef float    v8f  __attribute__((ext_vector_type(8)));
typedef float    f4   __attribute__((ext_vector_type(4)));

constexpr int   Bg   = 4096;
constexpr int   Pn   = 16;
constexpr int   EPGc = 192;
constexpr long  Nn   = (long)Bg * Pn;     // 65536 nodes
constexpr long  Ee   = (long)Bg * EPGc;   // 786432 edges

// softplus(x) - ln2 via hardware v_exp_f32 / v_log_f32 (native ocml)
__device__ inline float sp2f(float x) {
    float t = __expf(-fabsf(x));
    return fmaxf(x, 0.f) + __logf(1.f + t) - 0.69314718055994531f;
}
__device__ inline float sigmoid_fast(float x) {
    return __fdividef(1.f, 1.f + __expf(-x));
}
__device__ inline float tanh_fast(float x) {
    float e = __expf(-2.f * fabsf(x));
    float r = (1.f - e) * __fdividef(1.f, 1.f + e);
    return copysignf(r, x);
}

// ---------------- WMMA fragment helpers (CDNA5 16x16x32 f16 layouts) ----------

// A-matrix 16x32 fragment from row-major f16 LDS tile (stride in halfs).
// lane<16: row M=lane, K = {kt*32+0..7, kt*32+16..23}
// lane>=16: row M=lane-16, K = {kt*32+8..15, kt*32+24..31}
__device__ inline v16h ldsA_frag(const f16* t, int stride, int kt) {
    int lane = threadIdx.x & 31;
    int m = lane & 15;
    int kb = kt * 32 + ((lane & 16) ? 8 : 0);
    const h8* p0 = (const h8*)(t + m * stride + kb);
    const h8* p1 = (const h8*)(t + m * stride + kb + 16);
    h8 a = *p0, b = *p1;
    v16h r;
#pragma unroll
    for (int j = 0; j < 8; ++j) { r[j] = a[j]; r[8 + j] = b[j]; }
    return r;
}

// B-matrix 32x16 fragment from global WT[Npad][Kpad] (row-major, f16).
// lane holds column N = nt*16 + (lane&15); lane<16: K = {0..7, 16..23}, lane>=16: K = {8..15, 24..31}.
__device__ inline v16h ldB_frag(const f16* wt, int kpad, int nt, int kt) {
    int lane = threadIdx.x & 31;
    int n = nt * 16 + (lane & 15);
    int kb = kt * 32 + ((lane & 16) ? 8 : 0);
    const h8* p0 = (const h8*)(wt + (long)n * kpad + kb);
    const h8* p1 = (const h8*)(wt + (long)n * kpad + kb + 16);
    h8 a = *p0, b = *p1;
    v16h r;
#pragma unroll
    for (int j = 0; j < 8; ++j) { r[j] = a[j]; r[8 + j] = b[j]; }
    return r;
}

template <int KT, int NT>
__device__ inline void gemm(const f16* A, int As, const f16* WT, int Kpad, v8f* acc) {
#pragma unroll
    for (int nt = 0; nt < NT; ++nt)
#pragma unroll
        for (int j = 0; j < 8; ++j) acc[nt][j] = 0.f;
#pragma unroll
    for (int kt = 0; kt < KT; ++kt) {
        v16h a = ldsA_frag(A, As, kt);
#pragma unroll
        for (int nt = 0; nt < NT; ++nt) {
            v16h b = ldB_frag(WT, Kpad, nt, kt);
            acc[nt] = __builtin_amdgcn_wmma_f32_16x16x32_f16(
                false, a, false, b, (short)0, acc[nt], false, false);
        }
    }
}

// Write sp2(acc + bias) as f16 into LDS tile H (C/D layout: VGPR r -> row r / r+8).
template <int NT>
__device__ inline void epi_lds(const v8f* acc, const float* bias, f16* H, int Hs, int Hcol) {
    int lane = threadIdx.x & 31;
    int nb = lane & 15;
    int mo = (lane & 16) ? 8 : 0;
#pragma unroll
    for (int nt = 0; nt < NT; ++nt) {
        int n = nt * 16 + nb;
        float bv = bias[n];
#pragma unroll
        for (int r = 0; r < 8; ++r)
            H[(r + mo) * Hs + Hcol + n] = (f16)sp2f(acc[nt][r] + bv);
    }
}

// Load 16 f32, convert, store 16 f16 (two b128 LDS stores)
__device__ inline void cvt16(f16* dst, const float* s) {
    f4 x0 = *(const f4*)s, x1 = *(const f4*)(s + 4);
    f4 x2 = *(const f4*)(s + 8), x3 = *(const f4*)(s + 12);
    h8 a, b;
    a[0]=(f16)x0[0]; a[1]=(f16)x0[1]; a[2]=(f16)x0[2]; a[3]=(f16)x0[3];
    a[4]=(f16)x1[0]; a[5]=(f16)x1[1]; a[6]=(f16)x1[2]; a[7]=(f16)x1[3];
    b[0]=(f16)x2[0]; b[1]=(f16)x2[1]; b[2]=(f16)x2[2]; b[3]=(f16)x2[3];
    b[4]=(f16)x3[0]; b[5]=(f16)x3[1]; b[6]=(f16)x3[2]; b[7]=(f16)x3[3];
    *(h8*)dst = a; *(h8*)(dst + 8) = b;
}

// Wave-cooperative load of a 16-row x 32-col f32 slab into an f16 LDS tile.
// 2 lanes per row, 16 cols each. rowof(r) gives the source row index.
template <class F>
__device__ inline void wload32(const float* src, int sstride, int scol, F rowof,
                               f16* A, int astride, int acol) {
    int lane = threadIdx.x & 31;
    int r = lane >> 1;
    int ch = (lane & 1) * 16;
    const float* s = src + (long)rowof(r) * sstride + scol + ch;
    cvt16(A + r * astride + acol + ch, s);
}

// ---------------- weight prep: f32 [K][N] -> f16 WT[Npad][Kpad] (zero-padded) --
__global__ void k_prep_wt(const float* W, f16* WT, int K, int N, int Kp, int Np) {
    int idx = blockIdx.x * blockDim.x + threadIdx.x;
    if (idx >= Kp * Np) return;
    int n = idx / Kp, k = idx % Kp;
    float v = (k < K && n < N) ? W[(long)k * N + n] : 0.f;
    WT[(long)n * Kp + k] = (f16)v;
}

// ---------------- encoders -----------------------------------------------------
__global__ void k_enc_edge(const float* dist, const float* W0, const float* b0,
                           const f16* WT1, const float* b1, float* e) {
    __shared__ __align__(16) f16 sH[4][16 * 64];
    int wave = threadIdx.x >> 5;
    long m0 = ((long)blockIdx.x * 4 + wave) * 16;
    if (m0 >= Ee) return;
    f16* H = sH[wave];
    int lane = threadIdx.x & 31;
    int r = lane >> 1, c0 = (lane & 1) * 32;
    float d = dist[m0 + r];
    float rb = __expf(-4.0f * d * d);         // exp(-d^2 / 0.5^2)
#pragma unroll
    for (int c = 0; c < 32; ++c)
        H[r * 64 + c0 + c] = (f16)sp2f(rb * W0[c0 + c] + b0[c0 + c]);
    v8f a[2];
    gemm<2, 2>(H, 64, WT1, 64, a);
    int nb = lane & 15, mo = (lane & 16) ? 8 : 0;
#pragma unroll
    for (int nt = 0; nt < 2; ++nt) {
        int n = nt * 16 + nb; float bv = b1[n];
#pragma unroll
        for (int rr = 0; rr < 8; ++rr)
            e[(m0 + rr + mo) * 32 + n] = sp2f(a[nt][rr] + bv);
    }
}

__global__ void k_enc_node(const int* nf, const float* emb,
                           const f16* WT0, const float* b0,
                           const f16* WT1, const float* b1, float* v) {
    __shared__ __align__(16) f16 sA[4][16 * 32];
    __shared__ __align__(16) f16 sH[4][16 * 64];
    __shared__ int snt[4][16];
    int wave = threadIdx.x >> 5;
    long m0 = ((long)blockIdx.x * 4 + wave) * 16;
    if (m0 >= Nn) return;
    f16* A = sA[wave]; f16* H = sH[wave];
    int lane = threadIdx.x & 31;
    if (lane < 16) {
        const int* row = nf + (m0 + lane) * 90;
        int bv = row[0], arg = 0;
        for (int i = 1; i < 89; ++i) { int x = row[i]; if (x > bv) { bv = x; arg = i; } }
        snt[wave][lane] = arg;
    }
    int r = lane >> 1, ch = (lane & 1) * 8;
    int nt0 = snt[wave][r];
    const float* s = emb + nt0 * 16 + ch;
    f4 x0 = *(const f4*)s, x1 = *(const f4*)(s + 4);
    h8 a, z;
    a[0]=(f16)x0[0]; a[1]=(f16)x0[1]; a[2]=(f16)x0[2]; a[3]=(f16)x0[3];
    a[4]=(f16)x1[0]; a[5]=(f16)x1[1]; a[6]=(f16)x1[2]; a[7]=(f16)x1[3];
#pragma unroll
    for (int j = 0; j < 8; ++j) z[j] = (f16)0.f;
    *(h8*)(A + r * 32 + ch) = a;
    *(h8*)(A + r * 32 + 16 + ch) = z;                  // zero pad K 16..31
    v8f a1[4]; gemm<1, 4>(A, 32, WT0, 32, a1); epi_lds<4>(a1, b0, H, 64, 0);
    v8f a2[2]; gemm<2, 2>(H, 64, WT1, 64, a2);
    int nb = lane & 15, mo = (lane & 16) ? 8 : 0;
#pragma unroll
    for (int nt = 0; nt < 2; ++nt) {
        int n = nt * 16 + nb; float bv = b1[n];
#pragma unroll
        for (int rr = 0; rr < 8; ++rr)
            v[(m0 + rr + mo) * 32 + n] = sp2f(a2[nt][rr] + bv);
    }
}

__global__ void k_enc_state(const int* nf, const float* sf, const float* semb,
                            const f16* WT0, const float* b0,
                            const f16* WT1, const float* b1, float* u) {
    __shared__ __align__(16) f16 sA[4][16 * 32];
    __shared__ __align__(16) f16 sH[4][16 * 64];
    int wave = threadIdx.x >> 5;
    long m0 = ((long)blockIdx.x * 4 + wave) * 16;
    if (m0 >= Bg) return;
    f16* A = sA[wave]; f16* H = sH[wave];
    int lane = threadIdx.x & 31;
    if (lane < 16) {
        long g = m0 + lane;
        int focus = 20;
        for (int i = 0; i < Pn; ++i) {
            int fl = nf[(g * Pn + i) * 90 + 89];
            if (fl > 0) { focus = i; break; }
        }
        f16* Ar = A + lane * 32;
#pragma unroll
        for (int c = 0; c < 8; ++c)  Ar[c]      = (f16)sf[g * 8 + c];
#pragma unroll
        for (int c = 0; c < 8; ++c)  Ar[8 + c]  = (f16)semb[focus * 8 + c];
#pragma unroll
        for (int c = 16; c < 32; ++c) Ar[c]     = (f16)0.f;   // zero pad
    }
    v8f a1[4]; gemm<1, 4>(A, 32, WT0, 32, a1); epi_lds<4>(a1, b0, H, 64, 0);
    v8f a2[2]; gemm<2, 2>(H, 64, WT1, 64, a2);
    int nb = lane & 15, mo = (lane & 16) ? 8 : 0;
#pragma unroll
    for (int nt = 0; nt < 2; ++nt) {
        int n = nt * 16 + nb; float bv = b1[n];
#pragma unroll
        for (int rr = 0; rr < 8; ++rr)
            u[(m0 + rr + mo) * 32 + n] = sp2f(a2[nt][rr] + bv);
    }
}

// ---------------- dense (32->32, sp2) precompute for v / u ---------------------
__global__ void k_dense_rows(const float* x, float* y, const f16* WT, const float* b,
                             long nrows) {
    __shared__ __align__(16) f16 sA[4][16 * 32];
    int wave = threadIdx.x >> 5;
    long m0 = ((long)blockIdx.x * 4 + wave) * 16;
    if (m0 >= nrows) return;
    f16* A = sA[wave];
    wload32(x, 32, 0, [&](int r) { return (int)(m0 + r); }, A, 32, 0);
    v8f acc[2]; gemm<1, 2>(A, 32, WT, 32, acc);
    int lane = threadIdx.x & 31, nb = lane & 15, mo = (lane & 16) ? 8 : 0;
#pragma unroll
    for (int nt = 0; nt < 2; ++nt) {
        int n = nt * 16 + nb; float bv = b[n];
#pragma unroll
        for (int r = 0; r < 8; ++r)
            y[(m0 + r + mo) * 32 + n] = sp2f(acc[nt][r] + bv);
    }
}

// ---------------- edge conv: [v_src|v_dst|e(densed)|u_g] 128 -> 64 -> 64 -> 32 -
__global__ void k_edge_conv(const float* vin, float* e, float* e_new, const float* uin,
                            const int* srcI, const int* dstI,
                            const f16* WTd, const float* bd,
                            const f16* WT0, const float* b0,
                            const f16* WT1, const float* b1,
                            const f16* WT2, const float* b2) {
    __shared__ __align__(16) f16 sA[4][16 * 128];
    __shared__ __align__(16) f16 sH[4][16 * 64];
    int wave = threadIdx.x >> 5;
    long m0 = ((long)blockIdx.x * 4 + wave) * 16;
    if (m0 >= Ee) return;
    f16* A = sA[wave]; f16* H = sH[wave];
    int lane = threadIdx.x & 31;
    wload32(vin, 32, 0, [&](int r) { return srcI[m0 + r]; },          A, 128, 0);
    wload32(vin, 32, 0, [&](int r) { return dstI[m0 + r]; },          A, 128, 32);
    wload32(e,   32, 0, [&](int r) { return (int)(m0 + r); },         A, 128, 64);
    wload32(uin, 32, 0, [&](int r) { return (int)((m0 + r) / EPGc); },A, 128, 96);
    if (WTd) {  // fused dense_edge (32->32, sp2), overwrite cols 64..95
        v8f accd[2]; gemm<1, 2>(A + 64, 128, WTd, 32, accd);
        epi_lds<2>(accd, bd, A, 128, 64);
    }
    v8f a1[4]; gemm<4, 4>(A, 128, WT0, 128, a1); epi_lds<4>(a1, b0, H, 64, 0);
    v8f a2[4]; gemm<2, 4>(H, 64,  WT1, 64,  a2); epi_lds<4>(a2, b1, H, 64, 0);
    v8f a3[2]; gemm<2, 2>(H, 64,  WT2, 64,  a3);
    int nb = lane & 15, mo = (lane & 16) ? 8 : 0;
#pragma unroll
    for (int nt = 0; nt < 2; ++nt) {
        int n = nt * 16 + nb; float bv = b2[n];
#pragma unroll
        for (int r = 0; r < 8; ++r) {
            long m = m0 + r + mo;
            float en = sp2f(a3[nt][r] + bv);
            e_new[m * 32 + n] = en;
            e[m * 32 + n] += en;                       // skip: e = e_new + e0
        }
    }
}

// ---------------- mean of incoming edges per node (deterministic order) --------
__global__ void k_ve(const float* e_new, const int* dstI, float* ve) {
    int g = blockIdx.x, t = threadIdx.x;               // 128 threads
    __shared__ int sd[EPGc];
    for (int j = t; j < EPGc; j += 128) sd[j] = dstI[(long)g * EPGc + j];
    __syncthreads();
    int node = t >> 3, f0 = (t & 7) * 4;
    int target = g * Pn + node;
    float a0 = 0, a1 = 0, a2 = 0, a3 = 0; int cnt = 0;
    for (int j = 0; j < EPGc; ++j) {
        if (sd[j] == target) {
            const float* p = e_new + ((long)g * EPGc + j) * 32 + f0;
            a0 += p[0]; a1 += p[1]; a2 += p[2]; a3 += p[3]; ++cnt;
        }
    }
    float inv = 1.0f / (float)(cnt > 1 ? cnt : 1);
    float* o = ve + (long)target * 32 + f0;
    o[0] = a0 * inv; o[1] = a1 * inv; o[2] = a2 * inv; o[3] = a3 * inv;
}

// ---------------- node conv: [v|ve|u_g] 96 -> 64 -> 64 -> 32 -------------------
__global__ void k_node_conv(const float* vin, const float* ve, const float* uin,
                            float* v, float* v_new,
                            const f16* WT0, const float* b0,
                            const f16* WT1, const float* b1,
                            const f16* WT2, const float* b2) {
    __shared__ __align__(16) f16 sA[4][16 * 96];
    __shared__ __align__(16) f16 sH[4][16 * 64];
    int wave = threadIdx.x >> 5;
    long m0 = ((long)blockIdx.x * 4 + wave) * 16;
    if (m0 >= Nn) return;
    f16* A = sA[wave]; f16* H = sH[wave];
    int lane = threadIdx.x & 31;
    wload32(vin, 32, 0, [&](int r) { return (int)(m0 + r); },        A, 96, 0);
    wload32(ve,  32, 0, [&](int r) { return (int)(m0 + r); },        A, 96, 32);
    wload32(uin, 32, 0, [&](int r) { return (int)((m0 + r) / Pn); }, A, 96, 64);
    v8f a1[4]; gemm<3, 4>(A, 96, WT0, 96, a1); epi_lds<4>(a1, b0, H, 64, 0);
    v8f a2[4]; gemm<2, 4>(H, 64, WT1, 64, a2); epi_lds<4>(a2, b1, H, 64, 0);
    v8f a3[2]; gemm<2, 2>(H, 64, WT2, 64, a3);
    int nb = lane & 15, mo = (lane & 16) ? 8 : 0;
#pragma unroll
    for (int nt = 0; nt < 2; ++nt) {
        int n = nt * 16 + nb; float bv = b2[n];
#pragma unroll
        for (int r = 0; r < 8; ++r) {
            long m = m0 + r + mo;
            float vn = sp2f(a3[nt][r] + bv);
            v_new[m * 32 + n] = vn;
            v[m * 32 + n] += vn;                       // skip: v = v_new + v0
        }
    }
}

// ---------------- per-graph means of v_new / e_new -----------------------------
__global__ void k_means(const float* v_new, const float* e_new, float* uv, float* ue) {
    int g = blockIdx.x, f = threadIdx.x;               // 32 threads
    float s = 0.f;
    for (int i = 0; i < Pn; ++i) s += v_new[((long)g * Pn + i) * 32 + f];
    uv[(long)g * 32 + f] = s * (1.0f / Pn);
    float s2 = 0.f;
    for (int j = 0; j < EPGc; ++j) s2 += e_new[((long)g * EPGc + j) * 32 + f];
    ue[(long)g * 32 + f] = s2 * (1.0f / EPGc);
}

// ---------------- state conv: [u|uv|ue] 96 -> 64 -> 64 -> 32 -------------------
__global__ void k_state_conv(const float* uin, const float* uv, const float* ue,
                             float* u,
                             const f16* WT0, const float* b0,
                             const f16* WT1, const float* b1,
                             const f16* WT2, const float* b2) {
    __shared__ __align__(16) f16 sA[4][16 * 96];
    __shared__ __align__(16) f16 sH[4][16 * 64];
    int wave = threadIdx.x >> 5;
    long m0 = ((long)blockIdx.x * 4 + wave) * 16;
    if (m0 >= Bg) return;
    f16* A = sA[wave]; f16* H = sH[wave];
    int lane = threadIdx.x & 31;
    wload32(uin, 32, 0, [&](int r) { return (int)(m0 + r); }, A, 96, 0);
    wload32(uv,  32, 0, [&](int r) { return (int)(m0 + r); }, A, 96, 32);
    wload32(ue,  32, 0, [&](int r) { return (int)(m0 + r); }, A, 96, 64);
    v8f a1[4]; gemm<3, 4>(A, 96, WT0, 96, a1); epi_lds<4>(a1, b0, H, 64, 0);
    v8f a2[4]; gemm<2, 4>(H, 64, WT1, 64, a2); epi_lds<4>(a2, b1, H, 64, 0);
    v8f a3[2]; gemm<2, 2>(H, 64, WT2, 64, a3);
    int nb = lane & 15, mo = (lane & 16) ? 8 : 0;
#pragma unroll
    for (int nt = 0; nt < 2; ++nt) {
        int n = nt * 16 + nb; float bv = b2[n];
#pragma unroll
        for (int r = 0; r < 8; ++r) {
            long m = m0 + r + mo;
            u[m * 32 + n] += sp2f(a3[nt][r] + bv);     // skip: u = u_new + u0
        }
    }
}

// ---------------- Set2Set (3 iters, LSTM d=32), one block per graph ------------
template <int ROWS>
__global__ void k_s2s(const float* x, const float* Wih, const float* bih,
                      const float* Whh, const float* bhh, float* out) {
    int g = blockIdx.x, t = threadIdx.x;               // 128 threads
    __shared__ float xs[ROWS * 32];
    __shared__ float q[64], hh[32], cc[32], gates[128], sc[ROWS], red[2];
    for (int i = t; i < ROWS * 32; i += 128) xs[i] = x[(long)g * ROWS * 32 + i];
    if (t < 64) q[t] = 0.f;
    if (t < 32) { hh[t] = 0.f; cc[t] = 0.f; }
    __syncthreads();
    for (int it = 0; it < 3; ++it) {
        float gs = bih[t] + bhh[t];
        for (int k = 0; k < 64; ++k) gs += q[k] * Wih[k * 128 + t];
        for (int k = 0; k < 32; ++k) gs += hh[k] * Whh[k * 128 + t];
        gates[t] = gs;
        __syncthreads();
        if (t < 32) {   // torch gate order i,f,g,o
            float ig = sigmoid_fast(gates[t]);
            float fg = sigmoid_fast(gates[32 + t]);
            float gg = tanh_fast(gates[64 + t]);
            float og = sigmoid_fast(gates[96 + t]);
            float c = fg * cc[t] + ig * gg;
            cc[t] = c;
            hh[t] = og * tanh_fast(c);
        }
        __syncthreads();
        for (int r = t; r < ROWS; r += 128) {
            float d = 0.f;
            for (int k = 0; k < 32; ++k) d += xs[r * 32 + k] * hh[k];
            sc[r] = d;
        }
        __syncthreads();
        if (t == 0) { float mx = -3.4e38f; for (int r = 0; r < ROWS; ++r) mx = fmaxf(mx, sc[r]); red[0] = mx; }
        __syncthreads();
        for (int r = t; r < ROWS; r += 128) sc[r] = __expf(sc[r] - red[0]);
        __syncthreads();
        if (t == 0) { float s = 0.f; for (int r = 0; r < ROWS; ++r) s += sc[r]; red[1] = s; }
        __syncthreads();
        if (t < 32) {
            float a = 0.f;
            for (int r = 0; r < ROWS; ++r) a += sc[r] * xs[r * 32 + t];
            q[32 + t] = __fdividef(a, red[1]);
            q[t] = hh[t];
        }
        __syncthreads();
    }
    if (t < 64) out[(long)g * 64 + t] = q[t];
}

// ---------------- out_proj: [nvec|evec|u] 160 -> 32 -> 16 -> 88 (last linear) --
__global__ void k_out(const float* nvec, const float* evec, const float* u,
                      const f16* WT0, const float* b0,
                      const f16* WT1, const float* b1,
                      const f16* WT2, const float* b2, float* out) {
    __shared__ __align__(16) f16 sA[4][16 * 160];
    __shared__ __align__(16) f16 sH[4][16 * 32];
    __shared__ __align__(16) f16 sH2[4][16 * 32];
    int wave = threadIdx.x >> 5;
    long m0 = ((long)blockIdx.x * 4 + wave) * 16;
    if (m0 >= Bg) return;
    f16* A = sA[wave]; f16* H = sH[wave]; f16* H2 = sH2[wave];
    int lane = threadIdx.x & 31;
    wload32(nvec, 64, 0,  [&](int r) { return (int)(m0 + r); }, A, 160, 0);
    wload32(nvec, 64, 32, [&](int r) { return (int)(m0 + r); }, A, 160, 32);
    wload32(evec, 64, 0,  [&](int r) { return (int)(m0 + r); }, A, 160, 64);
    wload32(evec, 64, 32, [&](int r) { return (int)(m0 + r); }, A, 160, 96);
    wload32(u,    32, 0,  [&](int r) { return (int)(m0 + r); }, A, 160, 128);
    v8f a1[2]; gemm<5, 2>(A, 160, WT0, 160, a1); epi_lds<2>(a1, b0, H, 32, 0);
    for (int i = lane; i < 16 * 32; i += 32) H2[i] = (f16)0.f;   // zero pad K 16..31
    v8f a2[1]; gemm<1, 1>(H, 32, WT1, 32, a2); epi_lds<1>(a2, b1, H2, 32, 0);
    v8f a3[6]; gemm<1, 6>(H2, 32, WT2, 32, a3);                  // WT2 padded 96x32
    int nb = lane & 15, mo = (lane & 16) ? 8 : 0;
#pragma unroll
    for (int nt = 0; nt < 6; ++nt) {
        int n = nt * 16 + nb;
        if (n < 88) {
            float bv = b2[n];
#pragma unroll
            for (int r = 0; r < 8; ++r)
                out[(m0 + r + mo) * 88 + n] = a3[nt][r] + bv;    // no activation
        }
    }
}

// ---------------- host side ----------------------------------------------------
// d_in leaf order: top-level dict insertion order, params as JAX pytree
// (dict keys alphabetical, lists/tuples in order):
//  0 node_feat  1 edge_dist  2 state_feat  3 src  4 dst
//  params: blocks[0]: conv_edge 5-10, conv_node 11-16, conv_state 17-22
//          blocks[1]: ce 23-28, cn 29-34, cs 35-40, dense_e 41-42, dense_n 43-44, dense_s 45-46
//          blocks[2]: ce 47-52, cn 53-58, cs 59-64, de 65-66, dn 67-68, ds 69-70
//          edge_enc 71-74 | edge_s2s: W_hh 75, W_ih 76, b_hh 77, b_ih 78
//          node_emb 79 | node_enc 80-83 | node_s2s: W_hh 84, W_ih 85, b_hh 86, b_ih 87
//          out_proj 88-93 | state_emb 94 | state_enc 95-98
extern "C" void kernel_launch(void* const* d_in, const int* in_sizes, int n_in,
                              void* d_out, int out_size, void* d_ws, size_t ws_size,
                              hipStream_t stream) {
    (void)in_sizes; (void)n_in; (void)out_size; (void)ws_size;
    const int*   nf   = (const int*)d_in[0];
    const float* dist = (const float*)d_in[1];
    const float* sf   = (const float*)d_in[2];
    const int*   srcI = (const int*)d_in[3];
    const int*   dstI = (const int*)d_in[4];
    auto F = [&](int i) { return (const float*)d_in[i]; };

    size_t off = 0; char* base = (char*)d_ws;
    auto carve = [&](size_t bytes) -> void* {
        off = (off + 255) & ~(size_t)255;
        void* p = base + off; off += bytes; return p;
    };
    float* e    = (float*)carve(Ee * 32 * sizeof(float));
    float* e_nw = (float*)carve(Ee * 32 * sizeof(float));
    float* v    = (float*)carve(Nn * 32 * sizeof(float));
    float* v_nw = (float*)carve(Nn * 32 * sizeof(float));
    float* ve   = (float*)carve(Nn * 32 * sizeof(float));
    float* vd   = (float*)carve(Nn * 32 * sizeof(float));
    float* u    = (float*)carve((size_t)Bg * 32 * sizeof(float));
    float* ud   = (float*)carve((size_t)Bg * 32 * sizeof(float));
    float* uv   = (float*)carve((size_t)Bg * 32 * sizeof(float));
    float* ue   = (float*)carve((size_t)Bg * 32 * sizeof(float));
    float* nvec = (float*)carve((size_t)Bg * 64 * sizeof(float));
    float* evec = (float*)carve((size_t)Bg * 64 * sizeof(float));

    auto prep = [&](const float* W, int K, int N, int Kp, int Np) -> f16* {
        f16* wt = (f16*)carve((size_t)Kp * Np * sizeof(f16));
        int tot = Kp * Np;
        k_prep_wt<<<(tot + 255) / 256, 256, 0, stream>>>(W, wt, K, N, Kp, Np);
        return wt;
    };

    // encoders + out_proj
    f16* ee1 = prep(F(73), 64, 32, 64, 32);
    f16* ne0 = prep(F(80), 16, 64, 32, 64);
    f16* ne1 = prep(F(82), 64, 32, 64, 32);
    f16* se0 = prep(F(95), 16, 64, 32, 64);
    f16* se1 = prep(F(97), 64, 32, 64, 32);
    f16* op0 = prep(F(88), 160, 32, 160, 32);
    f16* op1 = prep(F(90), 32, 16, 32, 16);
    f16* op2 = prep(F(92), 16, 88, 32, 96);

    int ceb[3] = {5, 23, 47}, cnb[3] = {11, 29, 53}, csb[3] = {17, 35, 59};
    int deb[3] = {-1, 41, 65}, dnb[3] = {-1, 43, 67}, dsb[3] = {-1, 45, 69};
    f16 *ce0[3], *ce1[3], *ce2[3], *cn0[3], *cn1[3], *cn2[3],
        *cs0[3], *cs1[3], *cs2[3], *de[3], *dn[3], *dsw[3];
    for (int b = 0; b < 3; ++b) {
        ce0[b] = prep(F(ceb[b] + 0), 128, 64, 128, 64);
        ce1[b] = prep(F(ceb[b] + 2), 64, 64, 64, 64);
        ce2[b] = prep(F(ceb[b] + 4), 64, 32, 64, 32);
        cn0[b] = prep(F(cnb[b] + 0), 96, 64, 96, 64);
        cn1[b] = prep(F(cnb[b] + 2), 64, 64, 64, 64);
        cn2[b] = prep(F(cnb[b] + 4), 64, 32, 64, 32);
        cs0[b] = prep(F(csb[b] + 0), 96, 64, 96, 64);
        cs1[b] = prep(F(csb[b] + 2), 64, 64, 64, 64);
        cs2[b] = prep(F(csb[b] + 4), 64, 32, 64, 32);
        if (b > 0) {
            de[b]  = prep(F(deb[b]), 32, 32, 32, 32);
            dn[b]  = prep(F(dnb[b]), 32, 32, 32, 32);
            dsw[b] = prep(F(dsb[b]), 32, 32, 32, 32);
        } else { de[b] = nullptr; dn[b] = nullptr; dsw[b] = nullptr; }
    }

    // encoders
    k_enc_edge <<<Ee / 64, 128, 0, stream>>>(dist, F(71), F(72), ee1, F(74), e);
    k_enc_node <<<Nn / 64, 128, 0, stream>>>(nf, F(79), ne0, F(81), ne1, F(83), v);
    k_enc_state<<<Bg / 64, 128, 0, stream>>>(nf, sf, F(94), se0, F(96), se1, F(98), u);

    // 3 MEGNet blocks
    for (int b = 0; b < 3; ++b) {
        const float* vin = v; const float* uin = u;
        if (b > 0) {
            k_dense_rows<<<Nn / 64, 128, 0, stream>>>(v, vd, dn[b], F(dnb[b] + 1), Nn);
            k_dense_rows<<<Bg / 64, 128, 0, stream>>>(u, ud, dsw[b], F(dsb[b] + 1), (long)Bg);
            vin = vd; uin = ud;
        }
        k_edge_conv<<<Ee / 64, 128, 0, stream>>>(vin, e, e_nw, uin, srcI, dstI,
            de[b], (b > 0) ? F(deb[b] + 1) : nullptr,
            ce0[b], F(ceb[b] + 1), ce1[b], F(ceb[b] + 3), ce2[b], F(ceb[b] + 5));
        k_ve<<<Bg, 128, 0, stream>>>(e_nw, dstI, ve);
        k_node_conv<<<Nn / 64, 128, 0, stream>>>(vin, ve, uin, v, v_nw,
            cn0[b], F(cnb[b] + 1), cn1[b], F(cnb[b] + 3), cn2[b], F(cnb[b] + 5));
        k_means<<<Bg, 32, 0, stream>>>(v_nw, e_nw, uv, ue);
        k_state_conv<<<Bg / 64, 128, 0, stream>>>(uin, uv, ue, u,
            cs0[b], F(csb[b] + 1), cs1[b], F(csb[b] + 3), cs2[b], F(csb[b] + 5));
    }

    // Set2Set readouts (node: W_ih=85,b_ih=87,W_hh=84,b_hh=86; edge: 76,78,75,77)
    k_s2s<Pn>  <<<Bg, 128, 0, stream>>>(v, F(85), F(87), F(84), F(86), nvec);
    k_s2s<EPGc><<<Bg, 128, 0, stream>>>(e, F(76), F(78), F(75), F(77), evec);

    k_out<<<Bg / 64, 128, 0, stream>>>(nvec, evec, u, op0, F(89), op1, F(91),
                                       op2, F(93), (float*)d_out);
}
